// Depthawaregate_76733885710895
// MI455X (gfx1250) — compile-verified, matrix operand order checked by reading, not run on hardware
//
#include <hip/hip_runtime.h>
#include <hip/hip_bf16.h>
#include <math.h>

typedef __attribute__((ext_vector_type(16))) _Float16 v16h;
typedef __attribute__((ext_vector_type(8)))  float    v8f;

#define BN_  8
#define HH   128
#define WW   128
#define CC   128
#define CD   64
#define NPIX (BN_*HH*WW)          // 131072

// ---------------------------------------------------------------------------
// Kernel 1: per-pixel channel max / mean of fmapD [B,H,W,64] -> mxav [B,H,W,2]
// ---------------------------------------------------------------------------
__global__ void k_mxav(const float* __restrict__ fmapD, float* __restrict__ mxav) {
    int p = blockIdx.x * blockDim.x + threadIdx.x;
    if (p >= NPIX) return;
    const float4* src = (const float4*)(fmapD + (size_t)p * CD);
    float mx = -INFINITY, sum = 0.f;
#pragma unroll
    for (int i = 0; i < CD / 4; ++i) {
        float4 v = src[i];
        mx = fmaxf(mx, fmaxf(fmaxf(v.x, v.y), fmaxf(v.z, v.w)));
        sum += v.x + v.y + v.z + v.w;
    }
    mxav[2 * p + 0] = mx;
    mxav[2 * p + 1] = sum * (1.f / CD);
}

// ---------------------------------------------------------------------------
// Kernel 2: 5x5 SAME conv over mxav [B,H,W,2] -> dbuf [B,H,W]
// ---------------------------------------------------------------------------
__global__ void k_saconv(const float* __restrict__ mxav,
                         const float* __restrict__ w_sa,
                         float* __restrict__ dbuf) {
    int p = blockIdx.x * blockDim.x + threadIdx.x;
    if (p >= NPIX) return;
    int b = p / (HH * WW);
    int r = p - b * HH * WW;
    int h = r / WW;
    int w = r - h * WW;
    float acc = 0.f;
#pragma unroll
    for (int i = 0; i < 5; ++i) {
        int y = h + i - 2;
        if (y < 0 || y >= HH) continue;
#pragma unroll
        for (int j = 0; j < 5; ++j) {
            int x = w + j - 2;
            if (x < 0 || x >= WW) continue;
            const float* px = mxav + (size_t)(((b * HH + y) * WW + x)) * 2;
            acc += px[0] * w_sa[(i * 5 + j) * 2 + 0]
                 + px[1] * w_sa[(i * 5 + j) * 2 + 1];
        }
    }
    dbuf[p] = acc;
}

// ---------------------------------------------------------------------------
// Kernel 3: pack w_conv [3,3,C,C] f32 -> f16 WMMA B fragments
// layout: wpack[(kykx*4 + chunk)*8 + nt][lane(32)][j(16)]
// B element (k in 0..31 of chunk, n in 0..15 of ntile):
//   lane = n + ((k>>3)&1)*16 ; j = ((k>>4)<<3) + (k&7)      (ISA 7.12.2 mirror)
// ---------------------------------------------------------------------------
__global__ void k_pack(const float* __restrict__ w_conv, _Float16* __restrict__ wpack) {
    int tid = blockIdx.x * blockDim.x + threadIdx.x;
    if (tid >= 9 * CC * CC) return;
    int co   = tid % CC;
    int c    = (tid / CC) % CC;
    int kykx = tid / (CC * CC);           // ky*3+kx (matches [3,3,C,C] linear order)
    int chunk = c >> 5;
    int kk    = c & 31;
    int nt    = co >> 4;
    int n     = co & 15;
    int lane  = n + (((kk >> 3) & 1) << 4);
    int j     = ((kk >> 4) << 3) + (kk & 7);
    size_t idx = ((((size_t)(kykx * 4 + chunk) * 8 + nt) * 32 + lane) * 16) + j;
    wpack[idx] = (_Float16)w_conv[tid];
}

// ---------------------------------------------------------------------------
// Kernel 4: fused gate + gathered-GEMM (WMMA f16 -> f32) + BN + ReLU + residual
// block = 256 thr (8 waves); tile = 16 pixels (M) x 128 out-channels (N)
// ---------------------------------------------------------------------------
__global__ void __launch_bounds__(256, 2)
k_main(const float* __restrict__ fmapS, const _Float16* __restrict__ wpack,
       const float* __restrict__ dbuf,
       const float* __restrict__ gamma, const float* __restrict__ beta,
       const float* __restrict__ mmean, const float* __restrict__ mvar,
       float* __restrict__ out) {
    __shared__ __align__(32) _Float16 sA[4][32][16];   // 4 K-chunks of 16x32 A tile

    const int tilesPerRow = WW / 16;                   // 8
    int blk = blockIdx.x;
    int ox0 = (blk % tilesPerRow) * 16;
    int oy  = (blk / tilesPerRow) % HH;
    int b   =  blk / (tilesPerRow * HH);

    int t    = threadIdx.x;
    int wave = t >> 5;          // N-tile 0..7
    int lane = t & 31;
    int m       = t >> 4;       // staging row 0..15
    int colbase = (t & 15) * 8; // 8 consecutive channels per staging thread

    v8f acc = {};

    for (int ky = 0; ky < 3; ++ky) {
        for (int kx = 0; kx < 3; ++kx) {
            // --- per-row gather coords + gate (TF reshape + stride-3 conv algebra) ---
            int ox = ox0 + m;
            int q  = 3 * WW * ky + 3 * ox + kx;
            int wq = q / 9;
            int p  = q - 9 * wq;
            int pi = p / 3;
            int pj = p - 3 * pi;
            int y  = oy + pi - 1;
            int x  = wq + pj - 1;
            float gate  = 0.f;
            int   base  = 0;
            bool  valid = (y >= 0) & (y < HH) & (x >= 0) & (x < WW);
            if (valid) {
                float dc = dbuf[(b * HH + oy) * WW + wq];
                float pd = dbuf[(b * HH + y)  * WW + x];
                gate = __expf(-2.f * fabsf(pd - dc));
                base = ((b * HH + y) * WW + x) * CC;
            }
            float4 v0 = {0.f, 0.f, 0.f, 0.f}, v1 = v0;
            if (valid) {
                const float4* src = (const float4*)(fmapS + base + colbase);
                v0 = src[0];
                v1 = src[1];
            }
            float vals[8] = {v0.x, v0.y, v0.z, v0.w, v1.x, v1.y, v1.z, v1.w};

            __syncthreads();   // previous WMMA consumed sA
            // store gated f16 values directly in WMMA A-operand layout
#pragma unroll
            for (int cc = 0; cc < 8; ++cc) {
                int c     = colbase + cc;
                int chunk = c >> 5;
                int kk    = c & 31;
                int ln    = m + (((kk >> 3) & 1) << 4);
                int j     = ((kk >> 4) << 3) + (kk & 7);
                sA[chunk][ln][j] = (_Float16)(vals[cc] * gate);
            }
            __syncthreads();

            int kykx = ky * 3 + kx;
#pragma unroll
            for (int chunk = 0; chunk < 4; ++chunk) {
                v16h a = *(const v16h*)(&sA[chunk][lane][0]);
                const v16h* wp = (const v16h*)(wpack +
                    ((((size_t)(kykx * 4 + chunk) * 8 + wave) * 32 + lane) * 16));
                v16h bb = *wp;
                acc = __builtin_amdgcn_wmma_f32_16x16x32_f16(
                    false, a, false, bb, (short)0, acc, false, false);
            }
        }
    }

    // --- epilogue: BN (moving stats) + ReLU + residual, per C/D VGPR layout ---
    int n  = lane & 15;
    int co = wave * 16 + n;
    int mb = (lane >> 4) * 8;
    float s  = gamma[co] * rsqrtf(mvar[co] + 1e-3f);
    float mu = mmean[co];
    float bt = beta[co];
#pragma unroll
    for (int r = 0; r < 8; ++r) {
        int mm = mb + r;
        size_t idx = ((size_t)((b * HH + oy) * WW + ox0 + mm)) * CC + co;
        float v = s * (acc[r] - mu) + bt;
        v = fmaxf(v, 0.f) + fmapS[idx];
        out[idx] = v;
    }
}

// ---------------------------------------------------------------------------
extern "C" void kernel_launch(void* const* d_in, const int* in_sizes, int n_in,
                              void* d_out, int out_size, void* d_ws, size_t ws_size,
                              hipStream_t stream) {
    const float* fmapD  = (const float*)d_in[0];
    const float* fmapS  = (const float*)d_in[1];
    const float* w_sa   = (const float*)d_in[2];
    const float* w_conv = (const float*)d_in[3];
    const float* gamma  = (const float*)d_in[4];
    const float* beta   = (const float*)d_in[5];
    const float* mmean  = (const float*)d_in[6];
    const float* mvar   = (const float*)d_in[7];
    float* out = (float*)d_out;

    char* ws = (char*)d_ws;
    float*     dbuf  = (float*)ws;                              // 512 KB
    float*     mxav  = (float*)(ws + (size_t)NPIX * 4);         // 1 MB
    _Float16*  wpack = (_Float16*)(ws + (size_t)NPIX * 4 + (size_t)NPIX * 8);

    k_mxav  <<<(NPIX + 255) / 256, 256, 0, stream>>>(fmapD, mxav);
    k_saconv<<<(NPIX + 255) / 256, 256, 0, stream>>>(mxav, w_sa, dbuf);
    k_pack  <<<(9 * CC * CC + 255) / 256, 256, 0, stream>>>(w_conv, wpack);
    k_main  <<<BN_ * HH * (WW / 16), 256, 0, stream>>>(
        fmapS, wpack, dbuf, gamma, beta, mmean, mvar, out);
}
